// LaEnergiaNoAparece_8881992368533
// MI455X (gfx1250) — matrix operand order repaired and checked by reading, not compiled, not run on hardware
//
#include <hip/hip_runtime.h>

// ---------------------------------------------------------------------------
// Problem constants
// ---------------------------------------------------------------------------
#define NN      169        // 13*13 nodes
#define NBATCH  131072     // B
#define NT      11         // output-node tiles of 16 (169 -> 176)
#define NS      6          // k-depth slices of 32 (169 -> 192)
#define SIGMA_F 5.67e-8f
#define TCENTER 300.0f

typedef __attribute__((ext_vector_type(16))) _Float16     v16h;
typedef __attribute__((ext_vector_type(8)))  float        v8f;
typedef __attribute__((ext_vector_type(8)))  unsigned int v8u;

// K is a 5-point stencil: K[m,n] != 0 only for |m-n| in {0,1,13}.
// Tile (t,s) covers m in [16t,16t+15], n in [32s,32s+31]; nonzero iff bands overlap.
__host__ __device__ constexpr bool tile_nz(int t, int s) {
  return (16 * t + 15 + 13 >= 32 * s) && (16 * t - 13 <= 32 * s + 31);
}

__host__ __device__ constexpr int count_frags() {
  int c = 0;
  for (int t = 0; t < NT; ++t)
    for (int s = 0; s < NS; ++s)
      if (tile_nz(t, s)) ++c;
  return c;
}
constexpr int NFRAG = count_frags();   // = 21

// Compact fragment index for nonzero (t,s); -1 otherwise. Fully folds after unroll.
__device__ __forceinline__ int frag_index(int t, int s) {
  if (!tile_nz(t, s)) return -1;
  int c = 0;
  for (int tt = 0; tt < NT; ++tt)
    for (int ss = 0; ss < NS; ++ss) {
      if (tt == t && ss == s) return c;
      if (tile_nz(tt, ss)) ++c;
    }
  return -1;
}

// Workspace layout (in dwords)
#define BFRAG_DW   (NFRAG * 256)          // 21 frags * 32 lanes * 8 dwords
#define ROWSUM_OFF BFRAG_DW               // 169 floats
#define PART_OFF   (BFRAG_DW + 256)       // 8192 wave partials
#define NWAVES     (NBATCH / 16)          // 8192 batch-tiles == waves

// ---------------------------------------------------------------------------
// Kernel 1: pack K^T into the WMMA B-operand per-lane layout (f16) + rowsum(K)
// B[k=n_local, j=m]: lane = ((k/8)&1)*16 + (j%16)... per 16-bit 32-deep layout:
//   lanes 0-15  : n_local 0-7,16-23,  col j = lane
//   lanes 16-31 : n_local 8-15,24-31, col j = lane-16
//   vgpr v, half h -> n_local = (v<4?0:16) + (lane>=16?8:0) + 2(v&3) + h
// ---------------------------------------------------------------------------
__global__ __launch_bounds__(256) void pack_kernel(const float* __restrict__ K,
                                                   unsigned int* __restrict__ bfrag,
                                                   float* __restrict__ rowsum) {
  // row sums of K (exact 0/1 up to f32 rounding; used for the T-centering fixup)
  for (int m = threadIdx.x; m < NN; m += blockDim.x) {
    float s = 0.0f;
    for (int n = 0; n < NN; ++n) s += K[m * NN + n];
    rowsum[m] = s;
  }
  // packed fragments
  for (int i = threadIdx.x; i < NFRAG * 256; i += blockDim.x) {
    int v    = i & 7;
    int lane = (i >> 3) & 31;
    int f    = i >> 8;
    // recover (t,s) for compact fragment f
    int t = 0, s = 0, c = 0;
    bool done = false;
    for (t = 0; t < NT && !done; ++t)
      for (s = 0; s < NS; ++s)
        if (tile_nz(t, s)) { if (c == f) { done = true; break; } ++c; }
    t -= 1;  // loop increments t once after break-out

    int m  = t * 16 + (lane & 15);
    int n0 = s * 32 + ((v < 4) ? 0 : 16) + ((lane >= 16) ? 8 : 0) + (v & 3) * 2;
    int n1 = n0 + 1;
    _Float16 h0 = (m < NN && n0 < NN) ? (_Float16)K[m * NN + n0] : (_Float16)0.0f;
    _Float16 h1 = (m < NN && n1 < NN) ? (_Float16)K[m * NN + n1] : (_Float16)0.0f;
    unsigned short u0 = __builtin_bit_cast(unsigned short, h0);
    unsigned short u1 = __builtin_bit_cast(unsigned short, h1);
    bfrag[i] = (unsigned int)u0 | ((unsigned int)u1 << 16);
  }
}

// ---------------------------------------------------------------------------
// Kernel 2: main fused kernel. 8 waves/block, one 16-batch tile per wave.
// D[b_tile 16 x m 176] = (T-300) x K^T via v_wmma_f32_16x16x32_f16,
// fused with radiation + heat loads + |.| partial reduction.
// ---------------------------------------------------------------------------
__global__ __launch_bounds__(256) void main_kernel(const float* __restrict__ T,
                                                   const float* __restrict__ heat,
                                                   const float* __restrict__ ifc,
                                                   const float* __restrict__ tenv,
                                                   const float* __restrict__ e_diag,
                                                   const unsigned int* __restrict__ bfrag,
                                                   const float* __restrict__ rowsum,
                                                   float* __restrict__ partials) {
  __shared__ unsigned int sB[NFRAG * 256];   // 21.5 KB: all K fragments
  for (int i = threadIdx.x; i < NFRAG * 256; i += 256) sB[i] = bfrag[i];
  __syncthreads();

  const int wave = threadIdx.x >> 5;
  const int lane = threadIdx.x & 31;
  const int bt   = blockIdx.x * 8 + wave;     // 0..8191
  const int b0   = bt * 16;
  const int lm   = lane & 15;
  const int hi   = lane >> 4;                 // 0 | 1

  // --- A fragments: 16 batches x 192 nodes (centered, f16), 6 slices ---
  // 16-bit A 16x32 layout: lane L (M = L%16), vgpr v, half h:
  //   K = (v<4?0:16) + (L>=16?8:0) + (v&3)*2 + h
  v16h afrag[NS];
  const float* Trow = T + (b0 + lm) * NN;
#pragma unroll
  for (int s = 0; s < NS; ++s) {
    v16h a;
#pragma unroll
    for (int v = 0; v < 8; ++v) {
      int nb = s * 32 + ((v < 4) ? 0 : 16) + (hi ? 8 : 0) + (v & 3) * 2;
      float t0 = (nb     < NN) ? (Trow[nb]     - TCENTER) : 0.0f;
      float t1 = (nb + 1 < NN) ? (Trow[nb + 1] - TCENTER) : 0.0f;
      a[2 * v]     = (_Float16)t0;
      a[2 * v + 1] = (_Float16)t1;
    }
    afrag[s] = a;
  }

  // Tenv^4 for the 8 batches this lane accumulates (D rows M = r + 8*hi)
  float tenv4[8];
#pragma unroll
  for (int r = 0; r < 8; ++r) {
    float tv  = tenv[b0 + r + hi * 8];
    float tv2 = tv * tv;
    tenv4[r]  = tv2 * tv2;
  }

  float lsum = 0.0f;
  const v8u* sBv = (const v8u*)sB;

#pragma unroll
  for (int t = 0; t < NT; ++t) {
    v8f acc = {0.f, 0.f, 0.f, 0.f, 0.f, 0.f, 0.f, 0.f};
#pragma unroll
    for (int s = 0; s < NS; ++s) {
      const int fi = frag_index(t, s);
      if (fi < 0) continue;                       // skip all-zero K tiles
      v8u  raw = sBv[fi * 32 + lane];             // ds_load_b128 x2
      v16h bf  = __builtin_bit_cast(v16h, raw);
      acc = __builtin_amdgcn_wmma_f32_16x16x32_f16(
          /*neg_a=*/false, afrag[s], /*neg_b=*/false, bf,
          /*c_mod=*/(short)0, acc, /*reuse_a=*/false, /*reuse_b=*/false);
    }
    // D layout: lane L -> column m = t*16 + L%16; vgpr r -> batch row r + 8*hi
    const int m = t * 16 + lm;
    if (m < NN) {
      const float base = TCENTER * rowsum[m];     // undo T-centering (K linear)
      const float se   = SIGMA_F * e_diag[m];
#pragma unroll
      for (int r = 0; r < 8; ++r) {
        const int idx = (b0 + r + hi * 8) * NN + m;
        float Tv  = T[idx];
        float T2  = Tv * Tv;
        float rad = se * (T2 * T2 - tenv4[r]);
        float q   = heat[idx] + ifc[idx];
        float ex  = acc[r] + base + rad - q;
        lsum += fabsf(ex);
      }
    }
  }

  // wave32 reduce -> one partial per wave (deterministic, no atomics)
#pragma unroll
  for (int off = 16; off > 0; off >>= 1) lsum += __shfl_xor(lsum, off, 32);
  if (lane == 0) partials[bt] = lsum;
}

// ---------------------------------------------------------------------------
// Kernel 3: deterministic final reduction -> mean
// ---------------------------------------------------------------------------
__global__ __launch_bounds__(256) void reduce_kernel(const float* __restrict__ partials,
                                                     float* __restrict__ out) {
  __shared__ float sh[256];
  float s = 0.0f;
  for (int i = threadIdx.x; i < NWAVES; i += 256) s += partials[i];
  sh[threadIdx.x] = s;
  __syncthreads();
  for (int w = 128; w > 0; w >>= 1) {
    if (threadIdx.x < w) sh[threadIdx.x] += sh[threadIdx.x + w];
    __syncthreads();
  }
  if (threadIdx.x == 0)
    out[0] = sh[0] * (1.0f / ((float)NBATCH * (float)NN));
}

// ---------------------------------------------------------------------------
// Host entry
// ---------------------------------------------------------------------------
extern "C" void kernel_launch(void* const* d_in, const int* in_sizes, int n_in,
                              void* d_out, int out_size, void* d_ws, size_t ws_size,
                              hipStream_t stream) {
  const float* T    = (const float*)d_in[0];  // outputs [B,13,13]
  const float* heat = (const float*)d_in[1];  // heaters [B,13,13]
  const float* ifc  = (const float*)d_in[2];  // interfaces [B,13,13]
  const float* tenv = (const float*)d_in[3];  // Tenv [B,1]
  const float* K    = (const float*)d_in[4];  // K [169,169]
  const float* e    = (const float*)d_in[5];  // e_diag [169]
  float* out = (float*)d_out;

  unsigned int* bfrag = (unsigned int*)d_ws;
  float* rowsum   = (float*)d_ws + ROWSUM_OFF;
  float* partials = (float*)d_ws + PART_OFF;

  pack_kernel<<<1, 256, 0, stream>>>(K, bfrag, rowsum);
  main_kernel<<<NWAVES / 8, 256, 0, stream>>>(T, heat, ifc, tenv, e,
                                              bfrag, rowsum, partials);
  reduce_kernel<<<1, 256, 0, stream>>>(partials, out);
}
